// NAMPenult_24507083391391
// MI455X (gfx1250) — matrix operand order repaired and checked by reading, not compiled, not run on hardware
//
#include <hip/hip_runtime.h>

// CDNA5 (gfx1250) fused NAM per-feature MLP:
//   relu(x*W1+b1) -> (BxH)@(HxH) f16 WMMA -> relu(+b2) -> dot(W3)+b3
// W2 held in registers as WMMA B-fragments across a rolled 8-tile batch loop;
// layer-3 reduction done with DPP butterfly adds (no LDS traffic).

typedef __attribute__((ext_vector_type(16))) _Float16 v16h;
typedef __attribute__((ext_vector_type(8)))  float    v8f;

namespace {
constexpr int kF = 128;                               // features
constexpr int kH = 64;                                // hidden width
constexpr int kRowsPerWave = 16;                      // M per wave per iter
constexpr int kWaves = 8;                             // 256 threads
constexpr int kRowsPerBlock = kRowsPerWave * kWaves;  // 128 rows / iter
constexpr int kIters = 8;                             // batch sub-tiles / block
constexpr int kRowsPerBlockTotal = kRowsPerBlock * kIters;  // 1024
constexpr int kW2Stride = kH + 2;                     // LDS pad vs bank conflicts

// Butterfly-sum step over 16 lanes via DPP (folds into v_add_f32_dpp).
template <int CTRL>
__device__ __forceinline__ float dppAdd(float v) {
  int s = __builtin_bit_cast(int, v);
  int p = __builtin_amdgcn_update_dpp(0, s, CTRL, 0xF, 0xF, true);
  return v + __builtin_bit_cast(float, p);
}
__device__ __forceinline__ float rowReduce16(float v) {
  v = dppAdd<0xB1>(v);   // quad_perm(1,0,3,2)  : xor 1
  v = dppAdd<0x4E>(v);   // quad_perm(2,3,0,1)  : xor 2
  v = dppAdd<0x141>(v);  // row_half_mirror     : swap quads in 8-group
  v = dppAdd<0x140>(v);  // row_mirror          : swap 8-groups in row
  return v;
}
}  // namespace

__global__ __launch_bounds__(kWaves * 32)
void nam_mlp_wmma(const float* __restrict__ X,   // (B, F)
                  const float* __restrict__ W1,  // (F, H)
                  const float* __restrict__ B1,  // (F, H)
                  const float* __restrict__ W2,  // (F, H, H) (h=K, k=N) row-major
                  const float* __restrict__ B2,  // (F, H)
                  const float* __restrict__ W3,  // (F, H)
                  const float* __restrict__ B3,  // (F,)
                  float* __restrict__ Out)       // (B, F)
{
  __shared__ _Float16 sW2t[kH * kW2Stride];      // transposed: [n][k], f16
  __shared__ float sW1[kH], sB1[kH], sB2[kH], sW3[kH];

  const int f = blockIdx.x % kF;
  const int rowBase = (blockIdx.x / kF) * kRowsPerBlockTotal;
  const int tid = threadIdx.x;

  if (tid < kH) {
    sW1[tid] = W1[f * kH + tid];
    sB1[tid] = B1[f * kH + tid];
    sB2[tid] = B2[f * kH + tid];
    sW3[tid] = W3[f * kH + tid];
  }
  const float* W2f = W2 + (size_t)f * kH * kH;
  for (int i = tid; i < kH * kH; i += kWaves * 32) {
    const int h = i >> 6;        // GEMM K index
    const int k = i & 63;        // GEMM N index
    sW2t[k * kW2Stride + h] = (_Float16)W2f[i];
  }
  __syncthreads();

  const int lane  = tid & 31;
  const int wave  = tid >> 5;
  const int l16   = lane & 15;
  const int lhalf = lane >> 4;

  // ---- hoist W2 B-fragments into registers (loop-invariant, 64 VGPRs) ----
  v16h bm[4][2];
#pragma unroll
  for (int t = 0; t < 4; ++t) {
    const int n = 16 * t + l16;  // B layout: N = lane%16
#pragma unroll
    for (int s = 0; s < 2; ++s) {
#pragma unroll
      for (int v = 0; v < 8; ++v) {
        // B layout: VGPR v, lane half -> K = 32s + 16*(v/4) + 8*half + 2*(v%4)
        const int kg = 32 * s + 16 * (v >> 2) + 8 * lhalf + 2 * (v & 3);
        bm[t][s][2 * v]     = sW2t[n * kW2Stride + kg];
        bm[t][s][2 * v + 1] = sW2t[n * kW2Stride + kg + 1];
      }
    }
  }

  // ---- hoist W1/b1 pre-swizzled into the f16 A-fragment layout ----
  v16h w1f[2], b1f[2];
#pragma unroll
  for (int s = 0; s < 2; ++s) {
#pragma unroll
    for (int v = 0; v < 8; ++v) {
      // A layout: VGPR v, lane half -> K = 32s + 16*(v/4) + 8*half + 2*(v%4)
      const int k0 = 32 * s + 16 * (v >> 2) + 8 * lhalf + 2 * (v & 3);
      w1f[s][2 * v]     = (_Float16)sW1[k0];
      w1f[s][2 * v + 1] = (_Float16)sW1[k0 + 1];
      b1f[s][2 * v]     = (_Float16)sB1[k0];
      b1f[s][2 * v + 1] = (_Float16)sB1[k0 + 1];
    }
  }

  // ---- hoist epilogue constants (C-layout column n = 16t + lane%16) ----
  float b2v[4], w3v[4];
#pragma unroll
  for (int t = 0; t < 4; ++t) {
    b2v[t] = sB2[16 * t + l16];
    w3v[t] = sW3[16 * t + l16];
  }
  const float b3f = B3[f];

  v16h zeroh;
#pragma unroll
  for (int e = 0; e < 16; ++e) zeroh[e] = (_Float16)0.f;

  // ---- rolled batch loop: 8 x (128-row tile); B-frags stay in registers ----
  const size_t xStride = (size_t)kRowsPerBlock * kF;
  const float* xPtr = X + (size_t)(rowBase + wave * kRowsPerWave + l16) * kF + f;
  float xcur = xPtr[0];

#pragma unroll 1
  for (int it = 0; it < kIters; ++it) {
    // prefetch next tile's x while this tile computes
    float xnext = 0.f;
    if (it + 1 < kIters) xnext = xPtr[(size_t)(it + 1) * xStride];

    // layer 1 fused into the A-fragment build (packed f16 fma/max)
    const _Float16 xh = (_Float16)xcur;
    v16h xs;
#pragma unroll
    for (int e = 0; e < 16; ++e) xs[e] = xh;

    v16h a[2];
#pragma unroll
    for (int s = 0; s < 2; ++s) {
      v16h t = xs * w1f[s] + b1f[s];
      a[s] = __builtin_elementwise_max(t, zeroh);
    }

    // layer 2: 4 N-tiles x 2 K-steps of v_wmma_f32_16x16x32_f16
    v8f c[4];
#pragma unroll
    for (int t = 0; t < 4; ++t) { v8f z = {}; c[t] = z; }
#pragma unroll
    for (int t = 0; t < 4; ++t) {
#pragma unroll
      for (int s = 0; s < 2; ++s) {
        c[t] = __builtin_amdgcn_wmma_f32_16x16x32_f16(
            /*neg_a=*/false, a[s], /*neg_b=*/false, bm[t][s],
            /*c_mod=*/(short)0, c[t], /*reuse_a=*/false, /*reuse_b=*/false);
      }
    }

    // layer 3 fused epilogue: sum_n relu(c + b2[n]) * w3[n]
    float acc[8];
#pragma unroll
    for (int r = 0; r < 8; ++r) acc[r] = 0.f;
#pragma unroll
    for (int t = 0; t < 4; ++t) {
#pragma unroll
      for (int r = 0; r < 8; ++r) {
        float h2 = c[t][r] + b2v[t];
        h2 = h2 > 0.f ? h2 : 0.f;
        acc[r] = fmaf(h2, w3v[t], acc[r]);
      }
    }
    // 16-lane butterfly sums on the VALU (DPP), no LDS
#pragma unroll
    for (int r = 0; r < 8; ++r) acc[r] = rowReduce16(acc[r]);

    if (l16 == 0) {
      const int mBase = rowBase + it * kRowsPerBlock + wave * kRowsPerWave + 8 * lhalf;
#pragma unroll
      for (int r = 0; r < 8; ++r) {
        Out[(size_t)(mBase + r) * kF + f] = acc[r] + b3f;
      }
    }
    xcur = xnext;
  }
}

extern "C" void kernel_launch(void* const* d_in, const int* in_sizes, int n_in,
                              void* d_out, int out_size, void* d_ws, size_t ws_size,
                              hipStream_t stream) {
  const float* X  = (const float*)d_in[0];
  const float* W1 = (const float*)d_in[1];
  const float* B1 = (const float*)d_in[2];
  const float* W2 = (const float*)d_in[3];
  const float* B2 = (const float*)d_in[4];
  const float* W3 = (const float*)d_in[5];
  const float* B3 = (const float*)d_in[6];
  float* Out = (float*)d_out;

  const int Btotal = in_sizes[0] / kF;                     // 16384
  const int grid = (Btotal / kRowsPerBlockTotal) * kF;     // 16 * 128 = 2048
  nam_mlp_wmma<<<grid, kWaves * 32, 0, stream>>>(X, W1, B1, W2, B2, W3, B3, Out);
}
